// Tree_31430570672887
// MI455X (gfx1250) — matrix operand order repaired and checked by reading, not compiled, not run on hardware
//
#include <hip/hip_runtime.h>
#include <hip/hip_bf16.h>
#include <math.h>

// ---------------------------------------------------------------------------
// Types for CDNA5 WMMA (wave32): v_wmma_f32_16x16x32_bf16
// ---------------------------------------------------------------------------
typedef __bf16 bf16_t;
typedef __attribute__((ext_vector_type(16))) __bf16 v16bf;
typedef __attribute__((ext_vector_type(4)))  __bf16 v4bf;
typedef __attribute__((ext_vector_type(8)))  float  v8f;

#define TB   4096      // batch
#define TD   1024      // hidden dim
#define TC   1000      // classes

#define BM 128
#define BN 128
#define BK 32

// ---------------------------------------------------------------------------
// Tiled GEMM:  Y[M x N] = act(A[M x K](bf16) * W[K x N](f32->bf16) + bias)
// A tiles staged via async global->LDS (ASYNCcnt); W tiles staged through
// registers with packed f32->bf16 conversion.  f32 accumulation via WMMA.
// 256 threads = 8 waves; each wave computes a 32x64 tile (2x4 WMMA frags).
// LDS double-buffered, software pipelined, one barrier per k-step.
// grid = (ceil(N/128), M/128)
// ---------------------------------------------------------------------------
template <bool RELU, typename OutT>
__global__ __launch_bounds__(256)
void gemm_bias_act(const bf16_t* __restrict__ A, int lda,
                   const float* __restrict__ W, int ldw, int Nvalid,
                   const float* __restrict__ bias,
                   OutT* __restrict__ Y, int ldy, int K) {
    __shared__ bf16_t As[2][BM][BK];   // [m][k], k contiguous
    __shared__ bf16_t Bs[2][BN][BK];   // transposed: [n][k], k contiguous

    const int rowBase = blockIdx.y * BM;
    const int colBase = blockIdx.x * BN;
    const int tid  = threadIdx.x;
    const int lane = tid & 31;
    const int wid  = tid >> 5;
    const int waveM = wid & 3;      // 4 row-groups of 32
    const int waveN = wid >> 2;     // 2 col-groups of 64
    const int half  = lane >> 4;    // 0: lanes 0-15, 1: lanes 16-31
    const int mn    = lane & 15;

    // ---- A staging: async global->LDS, 16B chunks, 2 per thread ----
    // chunk = tid + p*256 over 512 chunks; r = chunk>>2, c = (chunk&3)*8
    const int ar0 = tid >> 2,         ac0 = (tid & 3) * 8;
    const int ar1 = (tid + 256) >> 2, ac1 = ac0;     // same (chunk&3)
    auto stageAsyncA = [&](int k0, int buf) {
        unsigned lds0 = (unsigned)(size_t)&As[buf][ar0][ac0];
        unsigned long long g0 =
            (unsigned long long)(size_t)(A + (size_t)(rowBase + ar0) * lda + (k0 + ac0));
        asm volatile("global_load_async_to_lds_b128 %0, %1, off"
                     :: "v"(lds0), "v"(g0) : "memory");
        unsigned lds1 = (unsigned)(size_t)&As[buf][ar1][ac1];
        unsigned long long g1 =
            (unsigned long long)(size_t)(A + (size_t)(rowBase + ar1) * lda + (k0 + ac1));
        asm volatile("global_load_async_to_lds_b128 %0, %1, off"
                     :: "v"(lds1), "v"(g1) : "memory");
    };

    // ---- W staging: registers + packed convert ----
    float bR[4][4];
    const int bn = tid & 127;        // n within tile
    const int bkg = tid >> 7;        // 0..1
    const bool bnValid = (colBase + bn) < Nvalid;
    auto stageLoadW = [&](int k0) {
        #pragma unroll
        for (int i = 0; i < 4; i++) {
            int k = (bkg * 4 + i) * 4;                // k multiple of 4
            #pragma unroll
            for (int j = 0; j < 4; j++)
                bR[i][j] = bnValid
                    ? W[(size_t)(k0 + k + j) * ldw + (colBase + bn)] : 0.0f;
        }
    };
    auto stageStoreW = [&](int buf) {
        #pragma unroll
        for (int i = 0; i < 4; i++) {
            int k = (bkg * 4 + i) * 4;
            v4bf v;
            #pragma unroll
            for (int j = 0; j < 4; j++) v[j] = (bf16_t)bR[i][j];
            *(v4bf*)&Bs[buf][bn][k] = v;              // ds_store_b64
        }
    };

    v8f acc[2][4] = {};

    stageAsyncA(0, 0);
    stageLoadW(0);
    stageStoreW(0);
    asm volatile("s_wait_asynccnt 0" ::: "memory");
    __syncthreads();

    int cur = 0;
    for (int k0 = 0; k0 < K; k0 += BK) {
        const bool hasNext = (k0 + BK) < K;
        if (hasNext) {
            stageAsyncA(k0 + BK, cur ^ 1);            // fire-and-forget to LDS
            stageLoadW(k0 + BK);                      // issue global loads early
        }

        // ---- fragments per ISA 16-bit layouts (vectorized LDS reads) ----
        // A 16x32: lane m = lane&15; VGPR0-3 hold K = 8*half..8*half+7,
        //          VGPR4-7 hold K = 16+8*half..+7  (two b128 loads)
        v16bf afr[2];
        #pragma unroll
        for (int mt = 0; mt < 2; mt++) {
            const bf16_t* ap0 = &As[cur][waveM * 32 + mt * 16 + mn][half * 8];
            const bf16_t* ap1 = &As[cur][waveM * 32 + mt * 16 + mn][16 + half * 8];
            #pragma unroll
            for (int e = 0; e < 8; e++) { afr[mt][e] = ap0[e]; afr[mt][8 + e] = ap1[e]; }
        }
        // B 32x16: lane n = lane&15; VGPR0-3 hold K = 8*half..8*half+7,
        //          VGPR4-7 hold K = 16+8*half..+7  (two b128 loads)
        v16bf bfr[4];
        #pragma unroll
        for (int nt = 0; nt < 4; nt++) {
            const bf16_t* bp0 = &Bs[cur][waveN * 64 + nt * 16 + mn][half * 8];
            const bf16_t* bp1 = &Bs[cur][waveN * 64 + nt * 16 + mn][16 + half * 8];
            #pragma unroll
            for (int e = 0; e < 8; e++) { bfr[nt][e] = bp0[e]; bfr[nt][8 + e] = bp1[e]; }
        }

        // ---- 8 WMMAs per wave per k-step (overlaps in-flight loads) ----
        #pragma unroll
        for (int mt = 0; mt < 2; mt++)
            #pragma unroll
            for (int nt = 0; nt < 4; nt++)
                acc[mt][nt] = __builtin_amdgcn_wmma_f32_16x16x32_bf16(
                    false, afr[mt], false, bfr[nt],
                    (short)0, acc[mt][nt], false, false);

        if (hasNext) stageStoreW(cur ^ 1);            // convert + LDS store
        asm volatile("s_wait_asynccnt 0" ::: "memory");
        __syncthreads();
        cur ^= 1;
    }

    // ---- epilogue: bias + optional ReLU.  D layout: m = v + 8*half, n = mn
    #pragma unroll
    for (int mt = 0; mt < 2; mt++) {
        #pragma unroll
        for (int nt = 0; nt < 4; nt++) {
            int col = colBase + waveN * 64 + nt * 16 + mn;
            if (col < Nvalid) {
                float bv = bias[col];
                #pragma unroll
                for (int v = 0; v < 8; v++) {
                    int row = rowBase + waveM * 32 + mt * 16 + v + 8 * half;
                    float val = acc[mt][nt][v] + bv;
                    if (RELU) val = fmaxf(val, 0.0f);
                    Y[(size_t)row * ldy + col] = (OutT)val;
                }
            }
        }
    }
}

// ---------------------------------------------------------------------------
// x (f32) -> bf16, one-time per launch.  4 elements per thread.
// ---------------------------------------------------------------------------
__global__ __launch_bounds__(256)
void f32_to_bf16_kernel(const float* __restrict__ in, bf16_t* __restrict__ out,
                        int n4) {
    int i = blockIdx.x * blockDim.x + threadIdx.x;
    if (i < n4) {
        float4 v = *(const float4*)(in + (size_t)i * 4);
        v4bf o;
        o[0] = (bf16_t)v.x; o[1] = (bf16_t)v.y;
        o[2] = (bf16_t)v.z; o[3] = (bf16_t)v.w;
        *(v4bf*)(out + (size_t)i * 4) = o;
    }
}

// ---------------------------------------------------------------------------
// Router: p = sigmoid(t[row] . W_r[node] + b_r[node]);
// r[2n+1] = r[n]*p ; r[2n+2] = r[n]*(1-p).   One wave per row.
// ---------------------------------------------------------------------------
__global__ __launch_bounds__(256)
void router_kernel(const bf16_t* __restrict__ t,
                   const float* __restrict__ wr,   // W_r + node*TD
                   const float* __restrict__ br,   // b_r + node
                   float* __restrict__ rbuf, int node) {
    int wid = threadIdx.x >> 5, lane = threadIdx.x & 31;
    int row = blockIdx.x * 8 + wid;
    const bf16_t* trow = t + (size_t)row * TD;
    float acc = 0.0f;
    #pragma unroll
    for (int j = 0; j < TD / 32; j++)
        acc += (float)trow[lane + 32 * j] * wr[lane + 32 * j];
    #pragma unroll
    for (int off = 16; off; off >>= 1)
        acc += __shfl_xor(acc, off, 32);
    if (lane == 0) {
        float p = 1.0f / (1.0f + expf(-(acc + br[0])));
        float parent = (node == 0) ? 1.0f : rbuf[(size_t)node * TB + row];
        rbuf[(size_t)(2 * node + 1) * TB + row] = parent * p;
        rbuf[(size_t)(2 * node + 2) * TB + row] = parent * (1.0f - p);
    }
}

// ---------------------------------------------------------------------------
// Per-row softmax of logits, scaled by reach prob, written/accumulated into y.
// One 256-thread block per row; C=1000 -> 4 cols per thread.
// ---------------------------------------------------------------------------
__global__ __launch_bounds__(256)
void softmax_acc_kernel(const float* __restrict__ logits, int ldl,
                        const float* __restrict__ rleaf,
                        float* __restrict__ y, int first) {
    __shared__ float redm[8];
    __shared__ float reds[8];
    int row = blockIdx.x;
    int t = threadIdx.x, wid = t >> 5, lane = t & 31;
    const float* lrow = logits + (size_t)row * ldl;

    float v[4];
    #pragma unroll
    for (int j = 0; j < 4; j++) {
        int c = t + j * 256;
        v[j] = (c < TC) ? lrow[c] : -3.0e38f;
    }
    float m = fmaxf(fmaxf(v[0], v[1]), fmaxf(v[2], v[3]));
    #pragma unroll
    for (int off = 16; off; off >>= 1) m = fmaxf(m, __shfl_xor(m, off, 32));
    if (lane == 0) redm[wid] = m;
    __syncthreads();
    m = redm[0];
    #pragma unroll
    for (int w = 1; w < 8; w++) m = fmaxf(m, redm[w]);

    float e[4]; float s = 0.0f;
    #pragma unroll
    for (int j = 0; j < 4; j++) {
        int c = t + j * 256;
        e[j] = (c < TC) ? expf(v[j] - m) : 0.0f;
        s += e[j];
    }
    #pragma unroll
    for (int off = 16; off; off >>= 1) s += __shfl_xor(s, off, 32);
    if (lane == 0) reds[wid] = s;
    __syncthreads();
    s = 0.0f;
    #pragma unroll
    for (int w = 0; w < 8; w++) s += reds[w];

    float scale = rleaf[row] / s;
    #pragma unroll
    for (int j = 0; j < 4; j++) {
        int c = t + j * 256;
        if (c < TC) {
            size_t o = (size_t)row * TC + c;
            if (first) y[o] = scale * e[j];
            else       y[o] += scale * e[j];
        }
    }
}

__global__ __launch_bounds__(256)
void log_final_kernel(float* __restrict__ y, int n) {
    int i = blockIdx.x * blockDim.x + threadIdx.x;
    if (i < n) y[i] = logf(1e-10f + y[i]);
}

// ---------------------------------------------------------------------------
// Launch.  Workspace (bf16 activations):
//   xbf + 7 activation slots (B x D bf16, 8 MB each), f32 logits slot,
//   reach-prob buffer.  ~81 MB total.
// ---------------------------------------------------------------------------
extern "C" void kernel_launch(void* const* d_in, const int* in_sizes, int n_in,
                              void* d_out, int out_size, void* d_ws, size_t ws_size,
                              hipStream_t stream) {
    const float* x   = (const float*)d_in[0];
    const float* W_t = (const float*)d_in[1];   // (15, D, D)
    const float* b_t = (const float*)d_in[2];   // (15, D)
    const float* W_r = (const float*)d_in[3];   // (7, D)
    const float* b_r = (const float*)d_in[4];   // (7,)
    const float* W_c = (const float*)d_in[5];   // (8, D, C)
    const float* b_c = (const float*)d_in[6];   // (8, C)
    float* out = (float*)d_out;                 // (B, C)

    const size_t SLOT = (size_t)TB * TD;        // elements per activation slot
    bf16_t* bws = (bf16_t*)d_ws;
    bf16_t* xbf = bws;                          // x in bf16
    bf16_t* slot[7];
    for (int i = 0; i < 7; i++) slot[i] = bws + (size_t)(1 + i) * SLOT;
    float* logits = (float*)(bws + (size_t)8 * SLOT);   // B x D f32 (ld=1024)
    float* rbuf = logits + SLOT;                // 15 * TB reach probs

    const dim3 gemmGrid(8, TB / BM);            // (8, 32)
    const dim3 blk(256);
    const dim3 routerGrid(TB / 8);

    auto transform = [&](const bf16_t* Ain, int node, bf16_t* outAct) {
        gemm_bias_act<true, bf16_t><<<gemmGrid, blk, 0, stream>>>(
            Ain, TD,
            W_t + (size_t)node * TD * TD, TD, TD,
            b_t + (size_t)node * TD,
            outAct, TD, TD);
    };
    auto router = [&](const bf16_t* tAct, int node) {
        router_kernel<<<routerGrid, blk, 0, stream>>>(
            tAct, W_r + (size_t)node * TD, b_r + node, rbuf, node);
    };

    // convert x once
    f32_to_bf16_kernel<<<dim3((TB * TD / 4 + 255) / 256), blk, 0, stream>>>(
        x, xbf, TB * TD / 4);

    // level 0
    transform(xbf, 0, slot[0]);
    router(slot[0], 0);
    // level 1
    transform(slot[0], 1, slot[1]);
    transform(slot[0], 2, slot[2]);
    router(slot[1], 1);
    router(slot[2], 2);
    // level 2
    transform(slot[1], 3, slot[3]);
    transform(slot[1], 4, slot[4]);
    transform(slot[2], 5, slot[5]);
    transform(slot[2], 6, slot[6]);
    router(slot[3], 3);
    router(slot[4], 4);
    router(slot[5], 5);
    router(slot[6], 6);
    // leaves
    for (int li = 0; li < 8; li++) {
        const int node = 7 + li;
        const bf16_t* parent = slot[3 + li / 2];
        transform(parent, node, slot[0]);                 // leaf activation
        gemm_bias_act<false, float><<<gemmGrid, blk, 0, stream>>>(
            slot[0], TD,
            W_c + (size_t)li * TD * TC, TC, TC,
            b_c + (size_t)li * TC,
            logits, TD, TD);                              // logits (ld=1024)
        softmax_acc_kernel<<<dim3(TB), blk, 0, stream>>>(
            logits, TD, rbuf + (size_t)node * TB, out, li == 0 ? 1 : 0);
    }
    // final log(1e-10 + y)
    const int n = TB * TC;
    log_final_kernel<<<dim3((n + 255) / 256), blk, 0, stream>>>(out, n);
}
